// GroupQueryAttention_4544075399207
// MI455X (gfx1250) — compile-verified, matrix operand order checked by reading, not run on hardware
//
#include <hip/hip_runtime.h>

// ---------- CDNA5 WMMA types ----------
typedef _Float16 v8h  __attribute__((ext_vector_type(8)));
typedef _Float16 v16h __attribute__((ext_vector_type(16)));
typedef float    v8f  __attribute__((ext_vector_type(8)));

// Load one 16-bit A/B fragment (16x32 / 32x16 f16) for this lane.
// Per ISA layout: lane holds 8 contiguous f16 at +0 and 8 more at +16 elements.
// Caller passes pointer already offset by (row*ld + k0 + kb), kb = 8*(lane>>4).
__device__ __forceinline__ v16h load_frag(const _Float16* p) {
    v8h lo = *(const v8h*)(p);
    v8h hi = *(const v8h*)(p + 16);
    return __builtin_shufflevector(lo, hi, 0,1,2,3,4,5,6,7,8,9,10,11,12,13,14,15);
}

__device__ __forceinline__ v8f wmma_f16(v16h a, v16h b, v8f c) {
    return __builtin_amdgcn_wmma_f32_16x16x32_f16(
        /*neg_a=*/false, a, /*neg_b=*/false, b,
        /*c_mod=*/(short)0, c, /*reuse_a=*/false, /*reuse_b=*/false);
}

// ---------- elementwise converts ----------
__global__ void cvt_f32_f16(const float* __restrict__ src, _Float16* __restrict__ dst, int n) {
    int i = blockIdx.x * blockDim.x + threadIdx.x;
    if (i < n) dst[i] = (_Float16)src[i];
}

// src [K,N] fp32 -> dst [N,K] f16 (weights transposed so B-fragments load contiguously)
__global__ void transpose_cvt(const float* __restrict__ src, _Float16* __restrict__ dst, int K, int N) {
    int i = blockIdx.x * blockDim.x + threadIdx.x;
    if (i < K * N) {
        int k = i / N, n = i % N;
        dst[(size_t)n * K + k] = (_Float16)src[i];
    }
}

// ---------- generic WMMA GEMM: C[M,N] = A[M,K] * B^T[N,K]^T + bias ----------
// One wave computes a 32(M) x 64(N) tile, K loop in steps of 32, software-
// pipelined: fragments for step i+1 are loaded before the WMMAs of step i so
// s_wait_loadcnt covers a full iteration of latency.
// mode 0: f16 store row-major (ldc)
// mode 1: f16 store transposed for V:  m=b*2048+sk, n=g*64+dk -> vt[(b*128+n)*2048+sk]
// mode 2: f32 store row-major (ldc)
__global__ void gemm16(const _Float16* __restrict__ A, const _Float16* __restrict__ BT,
                       const float* __restrict__ bias, void* __restrict__ Cout,
                       int M, int N, int K, int lda, int ldb, int ldc, int mode) {
    int wid  = (blockIdx.x * blockDim.x + threadIdx.x) >> 5;
    int lane = threadIdx.x & 31;
    int tilesN = N >> 6;
    int tilesM = M >> 5;
    if (wid >= tilesM * tilesN) return;
    int m0 = (wid / tilesN) << 5;
    int n0 = (wid % tilesN) << 6;

    int rloc = lane & 15;            // A: row within tile, B: col within subtile
    int kb   = (lane >> 4) << 3;     // k sub-offset for this lane half

    const _Float16* arow0 = A + (size_t)(m0 + rloc) * lda + kb;
    const _Float16* arow1 = arow0 + (size_t)16 * lda;
    const _Float16* brow  = BT + (size_t)(n0 + rloc) * ldb + kb;
    size_t bstep = (size_t)16 * ldb;

    v16h a0 = load_frag(arow0);
    v16h a1 = load_frag(arow1);
    v16h b0 = load_frag(brow);
    v16h b1 = load_frag(brow + bstep);
    v16h b2 = load_frag(brow + 2 * bstep);
    v16h b3 = load_frag(brow + 3 * bstep);

    v8f acc[2][4] = {};
    int klast = K - 32;
    for (int k0 = 0; k0 < klast; k0 += 32) {
        int kn = k0 + 32;
        // prefetch next k-step into fresh registers (issued before the waits
        // that guard this step's WMMAs)
        v16h a0n = load_frag(arow0 + kn);
        v16h a1n = load_frag(arow1 + kn);
        v16h b0n = load_frag(brow + kn);
        v16h b1n = load_frag(brow + bstep + kn);
        v16h b2n = load_frag(brow + 2 * bstep + kn);
        v16h b3n = load_frag(brow + 3 * bstep + kn);

        acc[0][0] = wmma_f16(a0, b0, acc[0][0]);
        acc[1][0] = wmma_f16(a1, b0, acc[1][0]);
        acc[0][1] = wmma_f16(a0, b1, acc[0][1]);
        acc[1][1] = wmma_f16(a1, b1, acc[1][1]);
        acc[0][2] = wmma_f16(a0, b2, acc[0][2]);
        acc[1][2] = wmma_f16(a1, b2, acc[1][2]);
        acc[0][3] = wmma_f16(a0, b3, acc[0][3]);
        acc[1][3] = wmma_f16(a1, b3, acc[1][3]);

        a0 = a0n; a1 = a1n; b0 = b0n; b1 = b1n; b2 = b2n; b3 = b3n;
    }
    // final (peeled) k-step
    acc[0][0] = wmma_f16(a0, b0, acc[0][0]);
    acc[1][0] = wmma_f16(a1, b0, acc[1][0]);
    acc[0][1] = wmma_f16(a0, b1, acc[0][1]);
    acc[1][1] = wmma_f16(a1, b1, acc[1][1]);
    acc[0][2] = wmma_f16(a0, b2, acc[0][2]);
    acc[1][2] = wmma_f16(a1, b2, acc[1][2]);
    acc[0][3] = wmma_f16(a0, b3, acc[0][3]);
    acc[1][3] = wmma_f16(a1, b3, acc[1][3]);

    int mloc = (lane >> 4) << 3;     // D layout: vgpr i -> row i (lo lanes) / 8+i (hi lanes)
#pragma unroll
    for (int hh = 0; hh < 2; ++hh) {
#pragma unroll
        for (int j = 0; j < 4; ++j) {
            int n = n0 + 16 * j + (lane & 15);
            float bv = bias ? bias[n] : 0.0f;
#pragma unroll
            for (int i = 0; i < 8; ++i) {
                int m = m0 + 16 * hh + mloc + i;
                float val = acc[hh][j][i] + bv;
                if (mode == 0) {
                    ((_Float16*)Cout)[(size_t)m * ldc + n] = (_Float16)val;
                } else if (mode == 1) {
                    int b  = m >> 11;
                    int sk = m & 2047;
                    ((_Float16*)Cout)[(((size_t)(b * 128 + n)) << 11) + sk] = (_Float16)val;
                } else {
                    ((float*)Cout)[(size_t)m * ldc + n] = val;
                }
            }
        }
    }
}

// ---------- flash GQA attention ----------
// One wave handles (b, h, 16 query rows). Computes S^T = Kp * Qp^T so that the
// softmax axis (sk) is the M dimension: per-column reductions are 8 in-lane ops
// + one shfl_xor(16). The D-layout of two stacked S^T tiles matches the
// B-fragment layout of the PV WMMA exactly (after f32->f16 pack).
// Pipelining: V-fragments for block kt are issued before the S^T WMMAs; the
// next block's K-fragments are issued before the softmax VALU work.
__global__ void flash_gqa(const _Float16* __restrict__ Qp,   // [b,sq,12,64] f16 (ld 768)
                          const _Float16* __restrict__ Kp,   // [b,sk, 2,64] f16 (ld 128)
                          const _Float16* __restrict__ Vt,   // [b, 2,64,sk] f16 (ld 2048)
                          _Float16* __restrict__ Oa)         // [b,sq,12,64] f16 (ld 768)
{
    int wid  = (blockIdx.x * blockDim.x + threadIdx.x) >> 5;
    int lane = threadIdx.x & 31;
    int bh  = wid >> 7;          // 48 (b,h) pairs
    int sqt = wid & 127;         // 128 query tiles of 16
    int b = bh / 12, h = bh % 12;
    int g = h & 1;               // jnp.tile pattern g0,g1,g0,g1,... -> h % GROUPS
    int sq0 = sqt << 4;

    int nloc = lane & 15;
    int kb   = (lane >> 4) << 3;

    // Q^T B-fragments (held for whole loop): n = sq row, contraction = dk (64 = 2 frags)
    const _Float16* qrow = Qp + (size_t)(b * 2048 + sq0 + nloc) * 768 + h * 64 + kb;
    v16h qb0 = load_frag(qrow);
    v16h qb1 = load_frag(qrow + 32);

    const _Float16* kbase = Kp + (size_t)(b * 2048) * 128 + g * 64 + kb;
    const _Float16* vbase = Vt + (((size_t)(b * 128 + g * 64)) << 11) + kb;

    float m_run = -3.0e38f, l_run = 0.0f;
    v8f o[4] = {};

    // preload K A-fragments for kt = 0
    const _Float16* kr0 = kbase + (size_t)nloc * 128;
    v16h a0 = load_frag(kr0);
    v16h a1 = load_frag(kr0 + 32);
    v16h a2 = load_frag(kr0 + (size_t)16 * 128);
    v16h a3 = load_frag(kr0 + (size_t)16 * 128 + 32);

    for (int kt = 0; kt < 2048; kt += 32) {
        // V A-fragments for THIS block: independent of S compute, issue early
        const _Float16* vr = vbase + kt;
        v16h va0 = load_frag(vr + (((size_t)(nloc)) << 11));
        v16h va1 = load_frag(vr + (((size_t)(16 + nloc)) << 11));
        v16h va2 = load_frag(vr + (((size_t)(32 + nloc)) << 11));
        v16h va3 = load_frag(vr + (((size_t)(48 + nloc)) << 11));

        // L2->WGP$ prefetch two key-blocks ahead
        __builtin_prefetch(kbase + (size_t)(kt + 64 + nloc) * 128, 0, 0);

        v8f z = {};
        v8f st1 = wmma_f16(a1, qb1, wmma_f16(a0, qb0, z));  // sk kt..kt+15
        v8f st2 = wmma_f16(a3, qb1, wmma_f16(a2, qb0, z));  // sk kt+16..kt+31

        // prefetch next block's K A-fragments (hidden behind softmax VALU)
        if (kt + 32 < 2048) {
            const _Float16* krn = kbase + (size_t)(kt + 32 + nloc) * 128;
            a0 = load_frag(krn);
            a1 = load_frag(krn + 32);
            a2 = load_frag(krn + (size_t)16 * 128);
            a3 = load_frag(krn + (size_t)16 * 128 + 32);
        }

        // online softmax per column (query row); reference applies NO 1/sqrt(dk)
        float t = st1[0];
#pragma unroll
        for (int i = 1; i < 8; ++i) t = fmaxf(t, st1[i]);
#pragma unroll
        for (int i = 0; i < 8; ++i) t = fmaxf(t, st2[i]);
        t = fmaxf(t, __shfl_xor(t, 16, 32));
        float m_new = fmaxf(m_run, t);
        float alpha = __expf(m_run - m_new);

        v16h pb;
        float rs = 0.0f;
#pragma unroll
        for (int i = 0; i < 8; ++i) {
            float p = __expf(st1[i] - m_new);
            pb[i] = (_Float16)p;  rs += p;
        }
#pragma unroll
        for (int i = 0; i < 8; ++i) {
            float p = __expf(st2[i] - m_new);
            pb[8 + i] = (_Float16)p;  rs += p;
        }
        rs += __shfl_xor(rs, 16, 32);
        l_run = l_run * alpha + rs;
        m_run = m_new;

#pragma unroll
        for (int r = 0; r < 4; ++r)
#pragma unroll
            for (int i = 0; i < 8; ++i) o[r][i] *= alpha;

        // O^T += V^T * P : A = V^T rows dk (contiguous sk), B = pb (already in-layout)
        o[0] = wmma_f16(va0, pb, o[0]);
        o[1] = wmma_f16(va1, pb, o[1]);
        o[2] = wmma_f16(va2, pb, o[2]);
        o[3] = wmma_f16(va3, pb, o[3]);
    }

    float inv = 1.0f / l_run;
    int mloc = (lane >> 4) << 3;
    _Float16* orow = Oa + (size_t)(b * 2048 + sq0 + nloc) * 768 + h * 64;
#pragma unroll
    for (int r = 0; r < 4; ++r)
#pragma unroll
        for (int i = 0; i < 8; ++i)
            orow[16 * r + mloc + i] = (_Float16)(o[r][i] * inv);
}

// ---------- launcher ----------
extern "C" void kernel_launch(void* const* d_in, const int* in_sizes, int n_in,
                              void* d_out, int out_size, void* d_ws, size_t ws_size,
                              hipStream_t stream) {
    const float* q  = (const float*)d_in[0];
    const float* k  = (const float*)d_in[1];
    const float* v  = (const float*)d_in[2];
    const float* Wq = (const float*)d_in[3];
    const float* bq = (const float*)d_in[4];
    const float* Wk = (const float*)d_in[5];
    const float* bk = (const float*)d_in[6];
    const float* Wv = (const float*)d_in[7];
    const float* bv = (const float*)d_in[8];
    const float* Wo = (const float*)d_in[9];
    const float* bo = (const float*)d_in[10];

    const int M   = 4 * 2048;        // 8192 rows
    const int nQK = M * 768;         // 6291456
    const int nW  = 768 * 768;       // 589824
    const int nWk = 768 * 128;       // 98304
    const int nKV = M * 128;         // 1048576

    _Float16* p = (_Float16*)d_ws;
    _Float16* q16 = p;            p += nQK;
    _Float16* k16 = p;            p += nQK;
    _Float16* v16 = p;            p += nQK;
    _Float16* wqT = p;            p += nW;
    _Float16* wkT = p;            p += nWk;
    _Float16* wvT = p;            p += nWk;
    _Float16* woT = p;            p += nW;
    _Float16* qp  = p;            p += nQK;
    _Float16* kp  = p;            p += nKV;
    _Float16* vt  = p;            p += nKV;
    _Float16* at  = p;            p += nQK;

    // Stage 1: converts
    cvt_f32_f16<<<(nQK + 255) / 256, 256, 0, stream>>>(q, q16, nQK);
    cvt_f32_f16<<<(nQK + 255) / 256, 256, 0, stream>>>(k, k16, nQK);
    cvt_f32_f16<<<(nQK + 255) / 256, 256, 0, stream>>>(v, v16, nQK);
    transpose_cvt<<<(nW  + 255) / 256, 256, 0, stream>>>(Wq, wqT, 768, 768);
    transpose_cvt<<<(nWk + 255) / 256, 256, 0, stream>>>(Wk, wkT, 768, 128);
    transpose_cvt<<<(nWk + 255) / 256, 256, 0, stream>>>(Wv, wvT, 768, 128);
    transpose_cvt<<<(nW  + 255) / 256, 256, 0, stream>>>(Wo, woT, 768, 768);

    // Stage 2: projections (WMMA). 32x64 tiles, 4 waves per 128-thread block.
    gemm16<<<(256 * 12) / 4, 128, 0, stream>>>(q16, wqT, bq, qp, M, 768, 768, 768, 768, 768, 0);
    gemm16<<<(256 *  2) / 4, 128, 0, stream>>>(k16, wkT, bk, kp, M, 128, 768, 768, 768, 128, 0);
    gemm16<<<(256 *  2) / 4, 128, 0, stream>>>(v16, wvT, bv, vt, M, 128, 768, 768, 768, 0, 1);

    // Stage 3: flash attention. 48 (b,h) * 128 sq-tiles = 6144 waves, 8 waves/block
    flash_gqa<<<6144 / 8, 256, 0, stream>>>(qp, kp, vt, at);

    // Stage 4: output projection -> fp32 d_out
    gemm16<<<(256 * 12) / 4, 128, 0, stream>>>(at, woT, bo, d_out, M, 768, 768, 768, 768, 768, 2);
}
